// ValueNetwork_50148038148299
// MI455X (gfx1250) — matrix operand order repaired and checked by reading, not compiled, not run on hardware
//
#include <hip/hip_runtime.h>
#include <hip/hip_bf16.h>

typedef __attribute__((ext_vector_type(16))) _Float16 v16h;
typedef __attribute__((ext_vector_type(8)))  _Float16 v8h;
typedef __attribute__((ext_vector_type(8)))  float    v8f;

// ---------------------------------------------------------------------------
// Geometry
// ---------------------------------------------------------------------------
#define NAGENTS   20
#define DIN       13
#define BPW       4            // batches per workgroup
#define ROWS      (BPW * NAGENTS)   // 80 rows -> 5 M-tiles
#define MTILES    5
#define NTHREADS  256
#define NWAVES    8

// LDS layout (bytes), all 16B aligned
#define OFF_BUF0   0            // 80 x 32  f16  (x, K-padded)           5120
#define OFF_BUF1   5120         // 80 x 160 f16  (h1)                   25600
#define OFF_BUF2   30720        // 80 x 128 f16  (h2, later s2)         20480
#define OFF_FEAT   51200        // 80 x 128 f16  (feat, kept)           20480
#define OFF_ATT    71680        // 80 x 224 f16  (att_in)               35840
#define OFF_S      107520       // 80 x 128 f16  (s1)                   20480
#define OFF_SC     128000       // 80 f32 scores                        320
#define OFF_WT     128320       // 80 f32 softmax weights               320
#define OFF_JOINT  128640       // 4 x 106 f32                          1696
#define OFF_V1     130336       // 4 x 150 f32                          2400
#define OFF_V2     132736       // 4 x 100 f32                          1600
#define OFF_V3     134336       // 4 x 100 f32                          1600
#define SMEM_BYTES 135936

// Packed-weight offsets in d_ws (f16 elements). Fragment = 32 lanes x 16 f16.
// frag index = nt*Ktiles + kt ; element order = frag*512 + lane*16 + i
#define W1P_OFF 0               // Kt=1, Nt=10 -> 10 frags  = 5120
#define W2P_OFF 5120            // Kt=5, Nt=8  -> 40 frags  = 20480
#define W3P_OFF 25600           // Kt=4, Nt=8  -> 32 frags  = 16384
#define A1P_OFF 41984           // Kt=7, Nt=8  -> 56 frags  = 28672
#define A2P_OFF 70656           // Kt=4, Nt=8  -> 32 frags  = 16384

// ---------------------------------------------------------------------------
// Pre-pass: fp32 weights -> f16, packed into WMMA B-fragment lane layout.
// B (KxN): lane L -> n = nt*16 + (L&15), khalf = L>>4 ; element i -> k =
// kt*32 + khalf*8 + (i&7) + (i>>3)*16 (same k order as the A fragment)
// ---------------------------------------------------------------------------
__global__ void pack_w_kernel(const float* __restrict__ w, _Float16* __restrict__ out,
                              int Kreal, int Nreal, int Ktiles, int total) {
    for (int idx = blockIdx.x * blockDim.x + threadIdx.x; idx < total;
         idx += gridDim.x * blockDim.x) {
        int frag   = idx >> 9;        // /512
        int within = idx & 511;
        int lane   = within >> 4;
        int i      = within & 15;
        int nt     = frag / Ktiles;
        int kt     = frag - nt * Ktiles;
        int n      = nt * 16 + (lane & 15);
        int k      = kt * 32 + ((lane >> 4) << 3) + (i & 7) + ((i >> 3) << 4);
        float v = (k < Kreal && n < Nreal) ? w[k * Nreal + n] : 0.0f;
        out[idx] = (_Float16)v;
    }
}

// ---------------------------------------------------------------------------
// One fused GEMM stage, nt-major: each wave owns an N-tile, loads its B
// fragments ONCE into registers, then sweeps all M-tiles reusing them
// (5x less L2 weight traffic than (mt,nt)-flat job order).
//
// A-fragment layout (16-bit A 16x32, ISA 7.12.2):
// lane L: m = mt*16 + (L&15), khalf = L>>4 ; elems 0..7 -> k = khalf*8 + i,
// elems 8..15 -> k = 16 + khalf*8 + (i-8)  => two contiguous 8xf16 LDS loads.
// C/D f32: VGPR r -> row mt*16 + r + 8*khalf, col nt*16 + (L&15).
// ---------------------------------------------------------------------------
template <int KTILES, int NTILES, int NREAL, bool RELU, bool DUAL>
__device__ __forceinline__ void gemm_stage(const _Float16* __restrict__ inB, int inStride,
                                           _Float16* __restrict__ outB, int outStride,
                                           _Float16* __restrict__ outB2, int out2Stride,
                                           int out2MaxN,
                                           const _Float16* __restrict__ wp,
                                           const float* __restrict__ bias,
                                           int wave, int lane) {
    const int lrow  = lane & 15;
    const int khalf = lane >> 4;
    for (int nt = wave; nt < NTILES; nt += NWAVES) {
        // --- load B fragments for this N-tile once (L2-resident stream) ---
        v16h bfrag[KTILES];
        const v16h* wf = (const v16h*)wp + (size_t)(nt * KTILES) * 32 + lane;
#pragma unroll
        for (int kt = 0; kt < KTILES; ++kt) bfrag[kt] = wf[kt * 32];

        const int   n  = nt * 16 + lrow;
        const float bv = (n < NREAL) ? bias[n] : 0.0f;

        for (int mt = 0; mt < MTILES; ++mt) {
            v8f c = {};
            int abase = (mt * 16 + lrow) * inStride + (khalf << 3);
#pragma unroll
            for (int kt = 0; kt < KTILES; ++kt) {
                union { v16h v; v8h h[2]; } u;
                u.h[0] = *(const v8h*)(inB + abase + kt * 32);
                u.h[1] = *(const v8h*)(inB + abase + kt * 32 + 16);
                c = __builtin_amdgcn_wmma_f32_16x16x32_f16(false, u.v, false, bfrag[kt],
                                                           (short)0, c, false, false);
            }
            int rbase = mt * 16 + (khalf << 3);
#pragma unroll
            for (int r = 0; r < 8; ++r) {
                float v = c[r] + bv;
                if (RELU) v = fmaxf(v, 0.0f);
                _Float16 hv = (n < NREAL) ? (_Float16)v : (_Float16)0.0f;
                outB[(rbase + r) * outStride + n] = hv;  // full padded stride -> zero pads
                if (DUAL && n < out2MaxN)
                    outB2[(rbase + r) * out2Stride + n] = hv;
            }
        }
    }
}

// ---------------------------------------------------------------------------
// Main fused kernel: one WG = 4 batches (80 agent rows), everything in LDS.
// ---------------------------------------------------------------------------
__global__ __launch_bounds__(NTHREADS) void value_net_kernel(
    const float* __restrict__ state,
    const float* __restrict__ b1, const float* __restrict__ b2, const float* __restrict__ b3,
    const float* __restrict__ ab1, const float* __restrict__ ab2,
    const float* __restrict__ a3, const float* __restrict__ ab3,
    const float* __restrict__ m1, const float* __restrict__ mb1,
    const float* __restrict__ m2, const float* __restrict__ mb2,
    const float* __restrict__ m3, const float* __restrict__ mb3,
    const float* __restrict__ m4, const float* __restrict__ mb4,
    const _Float16* __restrict__ w1p, const _Float16* __restrict__ w2p,
    const _Float16* __restrict__ w3p, const _Float16* __restrict__ a1p,
    const _Float16* __restrict__ a2p,
    float* __restrict__ out) {
    extern __shared__ char smem[];
    _Float16* buf0  = (_Float16*)(smem + OFF_BUF0);   // 80x32
    _Float16* buf1  = (_Float16*)(smem + OFF_BUF1);   // 80x160
    _Float16* buf2  = (_Float16*)(smem + OFF_BUF2);   // 80x128
    _Float16* featb = (_Float16*)(smem + OFF_FEAT);   // 80x128
    _Float16* attb  = (_Float16*)(smem + OFF_ATT);    // 80x224
    _Float16* sbuf  = (_Float16*)(smem + OFF_S);      // 80x128
    float* scoresL  = (float*)(smem + OFF_SC);
    float* wtsL     = (float*)(smem + OFF_WT);
    float* jointL   = (float*)(smem + OFF_JOINT);
    float* v1L      = (float*)(smem + OFF_V1);
    float* v2L      = (float*)(smem + OFF_V2);
    float* v3L      = (float*)(smem + OFF_V3);

    const int tid  = threadIdx.x;
    const int lane = tid & 31;
    const int wave = tid >> 5;
    const int b0   = blockIdx.x * BPW;

    // ---- stage 0: stage x (fp32 -> f16, K padded 13->32) + zero att tail ----
    for (int t = tid; t < ROWS * 32; t += NTHREADS) {
        int row = t >> 5, col = t & 31;
        float v = 0.0f;
        if (col < DIN) v = state[(size_t)(b0 * NAGENTS + row) * DIN + col];
        buf0[t] = (_Float16)v;
    }
    for (int t = tid; t < ROWS * 24; t += NTHREADS) {   // att_in cols 200..223 = 0
        int row = t / 24, col = 200 + t - row * 24;
        attb[row * 224 + col] = (_Float16)0.0f;
    }
    __syncthreads();

    // ---- L1: (80x13) @ (13x150), relu ----
    gemm_stage<1, 10, 150, true, false>(buf0, 32, buf1, 160, nullptr, 0, 0, w1p, b1, wave, lane);
    __syncthreads();
    // ---- L2: (80x150) @ (150x100), relu ----
    gemm_stage<5, 8, 100, true, false>(buf1, 160, buf2, 128, nullptr, 0, 0, w2p, b2, wave, lane);
    __syncthreads();
    // ---- L3: feat = (80x100) @ (100x100), linear; dual-write into att_in ----
    gemm_stage<4, 8, 100, false, true>(buf2, 128, featb, 128, attb, 224, 100, w3p, b3, wave, lane);
    __syncthreads();

    // ---- mean over agents -> att_in cols 100..199 ----
    for (int t = tid; t < BPW * 100; t += NTHREADS) {
        int bb = t / 100, d = t - bb * 100;
        float s = 0.0f;
        int base = (bb * NAGENTS) * 128 + d;
        for (int n = 0; n < NAGENTS; ++n) s += (float)featb[base + n * 128];
        _Float16 mh = (_Float16)(s * (1.0f / NAGENTS));
        int abase = (bb * NAGENTS) * 224 + 100 + d;
        for (int n = 0; n < NAGENTS; ++n) attb[abase + n * 224] = mh;
    }
    __syncthreads();

    // ---- A1: (80x200) @ (200x100), relu ----
    gemm_stage<7, 8, 100, true, false>(attb, 224, sbuf, 128, nullptr, 0, 0, a1p, ab1, wave, lane);
    __syncthreads();
    // ---- A2: (80x100) @ (100x100), relu (s2 overwrites dead h2) ----
    gemm_stage<4, 8, 100, true, false>(sbuf, 128, buf2, 128, nullptr, 0, 0, a2p, ab2, wave, lane);
    __syncthreads();

    // ---- scores = relu(s2 @ a3 + ab3) : 80 rows x dot(100), VALU ----
    for (int t = tid; t < ROWS; t += NTHREADS) {
        float s = ab3[0];
        for (int d = 0; d < 100; ++d) s += (float)buf2[t * 128 + d] * a3[d];
        scoresL[t] = fmaxf(s, 0.0f);
    }
    __syncthreads();

    // ---- softmax over the 20 agents, per batch ----
    for (int t = tid; t < BPW; t += NTHREADS) {
        float mx = -1e30f;
        for (int n = 0; n < NAGENTS; ++n) mx = fmaxf(mx, scoresL[t * NAGENTS + n]);
        float se = 0.0f;
        for (int n = 0; n < NAGENTS; ++n) {
            float e = __expf(scoresL[t * NAGENTS + n] - mx);
            wtsL[t * NAGENTS + n] = e;
            se += e;
        }
        float inv = 1.0f / se;
        for (int n = 0; n < NAGENTS; ++n) wtsL[t * NAGENTS + n] *= inv;
    }
    __syncthreads();

    // ---- weighted feature + self-state -> joint (4 x 106, f32) ----
    for (int t = tid; t < BPW * 100; t += NTHREADS) {
        int bb = t / 100, d = t - bb * 100;
        float s = 0.0f;
        for (int n = 0; n < NAGENTS; ++n)
            s += wtsL[bb * NAGENTS + n] * (float)featb[(bb * NAGENTS + n) * 128 + d];
        jointL[bb * 106 + 6 + d] = s;
    }
    for (int t = tid; t < BPW * 6; t += NTHREADS) {
        int bb = t / 6, d = t - bb * 6;
        jointL[bb * 106 + d] = state[(size_t)((b0 + bb) * NAGENTS) * DIN + d];
    }
    __syncthreads();

    // ---- final MLP on 4 rows (tiny: fp32 VALU) ----
    for (int t = tid; t < BPW * 150; t += NTHREADS) {
        int bb = t / 150, o = t - bb * 150;
        float s = mb1[o];
        for (int k = 0; k < 106; ++k) s += jointL[bb * 106 + k] * m1[k * 150 + o];
        v1L[bb * 150 + o] = fmaxf(s, 0.0f);
    }
    __syncthreads();
    for (int t = tid; t < BPW * 100; t += NTHREADS) {
        int bb = t / 100, o = t - bb * 100;
        float s = mb2[o];
        for (int k = 0; k < 150; ++k) s += v1L[bb * 150 + k] * m2[k * 100 + o];
        v2L[bb * 100 + o] = fmaxf(s, 0.0f);
    }
    __syncthreads();
    for (int t = tid; t < BPW * 100; t += NTHREADS) {
        int bb = t / 100, o = t - bb * 100;
        float s = mb3[o];
        for (int k = 0; k < 100; ++k) s += v2L[bb * 100 + k] * m3[k * 100 + o];
        v3L[bb * 100 + o] = fmaxf(s, 0.0f);
    }
    __syncthreads();
    for (int t = tid; t < BPW; t += NTHREADS) {
        float s = mb4[0];
        for (int k = 0; k < 100; ++k) s += v3L[t * 100 + k] * m4[k];
        out[b0 + t] = s;
    }
}

// ---------------------------------------------------------------------------
extern "C" void kernel_launch(void* const* d_in, const int* in_sizes, int n_in,
                              void* d_out, int out_size, void* d_ws, size_t ws_size,
                              hipStream_t stream) {
    const float* state = (const float*)d_in[0];
    const float* w1 = (const float*)d_in[1];  const float* b1  = (const float*)d_in[2];
    const float* w2 = (const float*)d_in[3];  const float* b2  = (const float*)d_in[4];
    const float* w3 = (const float*)d_in[5];  const float* b3  = (const float*)d_in[6];
    const float* a1 = (const float*)d_in[7];  const float* ab1 = (const float*)d_in[8];
    const float* a2 = (const float*)d_in[9];  const float* ab2 = (const float*)d_in[10];
    const float* a3 = (const float*)d_in[11]; const float* ab3 = (const float*)d_in[12];
    const float* m1 = (const float*)d_in[13]; const float* mb1 = (const float*)d_in[14];
    const float* m2 = (const float*)d_in[15]; const float* mb2 = (const float*)d_in[16];
    const float* m3 = (const float*)d_in[17]; const float* mb3 = (const float*)d_in[18];
    const float* m4 = (const float*)d_in[19]; const float* mb4 = (const float*)d_in[20];
    float* out = (float*)d_out;

    _Float16* ws  = (_Float16*)d_ws;
    _Float16* w1p = ws + W1P_OFF;
    _Float16* w2p = ws + W2P_OFF;
    _Float16* w3p = ws + W3P_OFF;
    _Float16* a1p = ws + A1P_OFF;
    _Float16* a2p = ws + A2P_OFF;

    // weight packing pre-pass (L2-resident afterwards; deterministic each call)
    auto packGrid = [](int total) { return (total + 255) / 256; };
    pack_w_kernel<<<packGrid(5120),  256, 0, stream>>>(w1, w1p, 13, 150, 1, 5120);
    pack_w_kernel<<<packGrid(20480), 256, 0, stream>>>(w2, w2p, 150, 100, 5, 20480);
    pack_w_kernel<<<packGrid(16384), 256, 0, stream>>>(w3, w3p, 100, 100, 4, 16384);
    pack_w_kernel<<<packGrid(28672), 256, 0, stream>>>(a1, a1p, 200, 100, 7, 28672);
    pack_w_kernel<<<packGrid(16384), 256, 0, stream>>>(a2, a2p, 100, 100, 4, 16384);

    int B = in_sizes[0] / (NAGENTS * DIN);   // 16384
    int grid = B / BPW;                      // 4096 workgroups
    value_net_kernel<<<grid, NTHREADS, SMEM_BYTES, stream>>>(
        state, b1, b2, b3, ab1, ab2, a3, ab3,
        m1, mb1, m2, mb2, m3, mb3, m4, mb4,
        w1p, w2p, w3p, a1p, a2p, out);
}
